// ONNBlock_30107720745323
// MI455X (gfx1250) — matrix operand order repaired and checked by reading, not compiled, not run on hardware
//
#include <hip/hip_runtime.h>
#include <hip/hip_bf16.h>
#include <stdint.h>

// ---------------------------------------------------------------------------
// Problem constants (from reference): B=64, n=128, T=n(n+1)/2=8256, H=1024,
// M(seeds)=1024 = (2n rotations) * (1 + 3 masks).
// ---------------------------------------------------------------------------
#define NB   64
#define NN   128
#define TT   8256
#define HH   1024
#define NSEED 1024

typedef __attribute__((ext_vector_type(8)))  _Float16 v8h;
typedef __attribute__((ext_vector_type(16))) _Float16 v16h;
typedef __attribute__((ext_vector_type(8)))  float    v8f;

// ---------------------------------------------------------------------------
// Kernel 1: transpose f32 weight (K x N, row-major) into two f16 arrays
// (N x K, row-major): hi = f16(w), lo = f16(w - f32(hi)).  The hi/lo split
// recovers ~f32 accuracy with two f16 WMMAs per K-step.
// ---------------------------------------------------------------------------
__global__ void onn_transpose_split(const float* __restrict__ W,
                                    _Float16* __restrict__ Thi,
                                    _Float16* __restrict__ Tlo,
                                    int K, int Nn) {
    int idx = blockIdx.x * 256 + threadIdx.x;
    int total = K * Nn;
    if (idx >= total) return;
    int k = idx / Nn;          // coalesced read across n
    int n = idx - k * Nn;
    float w = W[idx];
    _Float16 hi = (_Float16)w;
    _Float16 lo = (_Float16)(w - (float)hi);
    size_t o = (size_t)n * K + k;
    Thi[o] = hi;
    Tlo[o] = lo;
}

// ---------------------------------------------------------------------------
// Kernel 2: feature generation.  One block per batch row b (64 blocks x 256
// threads = 8 wave32s).  Each thread owns 4 seeds; the 128-bit state lives in
// 4 u32 words.  Per feature position we ballot across the wave and lane 0
// adds popcount(ballot) into the LDS counter -> 32x fewer LDS atomics.
// f[b][t] = count/512 - 1 (exact in f16).
// ---------------------------------------------------------------------------
__global__ void onn_features(const int* __restrict__ x,
                             const unsigned char* __restrict__ masks,
                             _Float16* __restrict__ f) {
    __shared__ unsigned cnt[TT];      // 33 KB
    __shared__ unsigned xw[4];        // packed x bits for this batch row
    __shared__ unsigned mw[3][4];     // packed mask bits (3 masks x 128)

    const int b   = blockIdx.x;
    const int tid = threadIdx.x;

    for (int t = tid; t < TT; t += 256) cnt[t] = 0u;
    if (tid < 4)  xw[tid] = 0u;
    if (tid < 12) ((unsigned*)mw)[tid] = 0u;
    __syncthreads();

    if (tid < NN) {
        if (x[b * NN + tid]) atomicOr(&xw[tid >> 5], 1u << (tid & 31));
    }
    if (tid < 3 * NN) {
        int r = tid >> 7, j = tid & 127;
        if (masks[tid]) atomicOr(&mw[r][j >> 5], 1u << (j & 31));
    }
    __syncthreads();

    for (int rep = 0; rep < NSEED / 256; ++rep) {
        const int m = rep * 256 + tid;       // seed index 0..1023
        const int s = m >> 2;                // rotation index 0..255
        const int c = m & 3;                 // 0 = unmasked, 1..3 = mask c-1

        // Build seed: rot for s<128, reversed-rot for s>=128, optional XOR mask.
        unsigned cur[4] = {0u, 0u, 0u, 0u};
        for (int j = 0; j < NN; ++j) {
            int src = (s < NN) ? ((s + j) & (NN - 1))
                               : (NN - 1 - ((s - NN + j) & (NN - 1)));
            unsigned bit = (xw[src >> 5] >> (src & 31)) & 1u;
            if (c) bit ^= (mw[c - 1][j >> 5] >> (j & 31)) & 1u;
            cur[j >> 5] |= bit << (j & 31);
        }

        // Evolution: row k has length 128-k; record then cur = XNOR(cur, cur>>1).
        int t = 0;
        int L = NN;
        for (int k = 0; k < NN; ++k) {
            for (int i = 0; i < L; ++i) {
                unsigned bit = (cur[i >> 5] >> (i & 31)) & 1u;
                unsigned bal = __builtin_amdgcn_ballot_w32(bit != 0u);
                if ((tid & 31) == 0)
                    atomicAdd(&cnt[t + i], (unsigned)__builtin_popcount(bal));
            }
            t += L;
            unsigned n0 = ~(cur[0] ^ ((cur[0] >> 1) | (cur[1] << 31)));
            unsigned n1 = ~(cur[1] ^ ((cur[1] >> 1) | (cur[2] << 31)));
            unsigned n2 = ~(cur[2] ^ ((cur[2] >> 1) | (cur[3] << 31)));
            unsigned n3 = ~(cur[3] ^  (cur[3] >> 1));
            cur[0] = n0; cur[1] = n1; cur[2] = n2; cur[3] = n3;
            --L;
        }
    }
    __syncthreads();

    for (int t = tid; t < TT; t += 256)
        f[(size_t)b * TT + t] = (_Float16)((float)cnt[t] * (1.0f / 512.0f) - 1.0f);
}

// ---------------------------------------------------------------------------
// Kernel 3: WMMA GEMM  C(M x Nn) = A(M x K, f16 row-major)
//                                @ B(K x Nn) given as BT(Nn x K, f16 hi/lo)
// then + bias, optional relu.  One 16x16 tile per wave, 8 waves per block.
// Grid: (Nn/16/8, M/16).  Uses v_wmma_f32_16x16x32_f16 with f32 accumulator;
// hi and lo weight planes accumulate into the same accumulator.
//
// VGPR layouts used here (contiguous k order, same for A and B):
//   A (16x32 f16): lane = m + 16g holds K = 16g..16g+7 (VGPR0-3) and
//                  K = 16g+8..16g+15 (VGPR4-7)  -> two b128 loads.
//   B (32x16 f16): lane = n + 16g holds K = 16g..16g+15 contiguously
//                  -> one 32-byte load from BT row n.
//   C/D 16x16 f32: lane l holds C[vr + 8*(l>>4)][l&15], vr = 0..7.
// ---------------------------------------------------------------------------
template<bool RELU, bool OUTF16>
__global__ void onn_wmma_gemm(const _Float16* __restrict__ A,
                              const _Float16* __restrict__ BThi,
                              const _Float16* __restrict__ BTlo,
                              const float* __restrict__ bias,
                              void* __restrict__ out,
                              int Nn, int K) {
    const int lane  = threadIdx.x & 31;
    const int wid   = threadIdx.x >> 5;
    const int ntile = blockIdx.x * 8 + wid;
    const int mtile = blockIdx.y;
    const int g = lane >> 4;
    const int r = lane & 15;

    const _Float16* arow  = A    + (size_t)(mtile * 16 + r) * K;
    const _Float16* brhi  = BThi + (size_t)(ntile * 16 + r) * K;
    const _Float16* brlo  = BTlo + (size_t)(ntile * 16 + r) * K;

    v8f acc = {};
    for (int k0 = 0; k0 < K; k0 += 32) {
        v8h alo8 = *(const v8h*)(arow + k0 + 16 * g);
        v8h ahi8 = *(const v8h*)(arow + k0 + 16 * g + 8);
        v16h a;
#pragma unroll
        for (int i = 0; i < 8; ++i) { a[i] = alo8[i]; a[i + 8] = ahi8[i]; }

        v16h bhi = *(const v16h*)(brhi + k0 + 16 * g);
        v16h blo = *(const v16h*)(brlo + k0 + 16 * g);

        acc = __builtin_amdgcn_wmma_f32_16x16x32_f16(
                  false, a, false, bhi, (short)0, acc, false, false);
        acc = __builtin_amdgcn_wmma_f32_16x16x32_f16(
                  false, a, false, blo, (short)0, acc, false, false);
    }

    const int ncol = ntile * 16 + r;
    const int m0   = mtile * 16 + 8 * g;
    const float bv = bias[ncol];
#pragma unroll
    for (int vr = 0; vr < 8; ++vr) {
        float v = acc[vr] + bv;
        if (RELU) v = v > 0.0f ? v : 0.0f;
        if (OUTF16)
            ((_Float16*)out)[(size_t)(m0 + vr) * Nn + ncol] = (_Float16)v;
        else
            ((float*)out)[(size_t)(m0 + vr) * Nn + ncol] = v;
    }
}

// ---------------------------------------------------------------------------
// Kernel 4: final matvec  out[b] = h1[b,:] . Wo + bo   (1024 -> 1)
// ---------------------------------------------------------------------------
__global__ void onn_out(const float* __restrict__ h1,
                        const float* __restrict__ Wo,
                        const float* __restrict__ bo,
                        float* __restrict__ out) {
    __shared__ float red[256];
    const int b = blockIdx.x, tid = threadIdx.x;
    float s = 0.0f;
    for (int i = tid; i < HH; i += 256) s += h1[(size_t)b * HH + i] * Wo[i];
    red[tid] = s;
    __syncthreads();
    for (int st = 128; st > 0; st >>= 1) {
        if (tid < st) red[tid] += red[tid + st];
        __syncthreads();
    }
    if (tid == 0) out[b] = red[0] + bo[0];
}

// ---------------------------------------------------------------------------
// Launch.  d_in order: x, masks, W0, b0, W1, b1, Wo, bo.
// Workspace layout (bytes, all 32B-aligned):
//   W0T_hi 16,908,288 | W0T_lo 16,908,288 | W1T_hi 2,097,152 | W1T_lo 2,097,152
//   f(f16) 1,056,768  | h0(f16) 131,072   | h1(f32) 262,144   = ~39.5 MB total
// ---------------------------------------------------------------------------
extern "C" void kernel_launch(void* const* d_in, const int* in_sizes, int n_in,
                              void* d_out, int out_size, void* d_ws, size_t ws_size,
                              hipStream_t stream) {
    (void)in_sizes; (void)n_in; (void)out_size; (void)ws_size;

    const int*           x     = (const int*)d_in[0];
    const unsigned char* masks = (const unsigned char*)d_in[1];
    const float*         W0    = (const float*)d_in[2];
    const float*         b0    = (const float*)d_in[3];
    const float*         W1    = (const float*)d_in[4];
    const float*         b1    = (const float*)d_in[5];
    const float*         Wo    = (const float*)d_in[6];
    const float*         bo    = (const float*)d_in[7];
    float*               out   = (float*)d_out;

    char* ws = (char*)d_ws;
    const size_t szW0T = (size_t)HH * TT * sizeof(_Float16);   // 16,908,288
    const size_t szW1T = (size_t)HH * HH * sizeof(_Float16);   //  2,097,152
    _Float16* W0Thi = (_Float16*)(ws);
    _Float16* W0Tlo = (_Float16*)(ws + szW0T);
    _Float16* W1Thi = (_Float16*)(ws + 2 * szW0T);
    _Float16* W1Tlo = (_Float16*)(ws + 2 * szW0T + szW1T);
    _Float16* fbuf  = (_Float16*)(ws + 2 * szW0T + 2 * szW1T);
    _Float16* h0    = (_Float16*)(ws + 2 * szW0T + 2 * szW1T
                                     + (size_t)NB * TT * sizeof(_Float16));
    float*    h1    = (float*)   (ws + 2 * szW0T + 2 * szW1T
                                     + (size_t)NB * TT * sizeof(_Float16)
                                     + (size_t)NB * HH * sizeof(_Float16));

    // 1) weight transpose + hi/lo split
    {
        int total = TT * HH;
        onn_transpose_split<<<(total + 255) / 256, 256, 0, stream>>>(
            W0, W0Thi, W0Tlo, TT, HH);
    }
    {
        int total = HH * HH;
        onn_transpose_split<<<(total + 255) / 256, 256, 0, stream>>>(
            W1, W1Thi, W1Tlo, HH, HH);
    }

    // 2) bit-evolution features -> f16 (exact)
    onn_features<<<NB, 256, 0, stream>>>(x, masks, fbuf);

    // 3) GEMM1: h0 = relu(f @ W0 + b0)   M=64, N=1024, K=8256
    {
        dim3 grid(HH / 16 / 8, NB / 16);
        onn_wmma_gemm<true, true><<<grid, 256, 0, stream>>>(
            fbuf, W0Thi, W0Tlo, b0, (void*)h0, HH, TT);
    }

    // 4) GEMM2: h1 = relu(h0 @ W1 + b1)  M=64, N=1024, K=1024
    {
        dim3 grid(HH / 16 / 8, NB / 16);
        onn_wmma_gemm<true, false><<<grid, 256, 0, stream>>>(
            h0, W1Thi, W1Tlo, b1, (void*)h1, HH, HH);
    }

    // 5) out = h1 @ Wo + bo
    onn_out<<<NB, 256, 0, stream>>>(h1, Wo, bo, out);
}
